// VisionTransformerBlock_41644002902525
// MI455X (gfx1250) — matrix operand order repaired and checked by reading, not compiled, not run on hardware
//
#include <hip/hip_runtime.h>
#include <math.h>

#define EMBED 768
#define HIDDEN 3072
#define SEQ 577
#define BATCH 32
#define NTOK (BATCH * SEQ)
#define SP 608  // SEQ padded to a multiple of 32

typedef __bf16 bf16;
typedef __attribute__((ext_vector_type(16))) __bf16 v16bf;
typedef __attribute__((ext_vector_type(8)))  __bf16 v8bf;
typedef __attribute__((ext_vector_type(8)))  float  v8f;
typedef int v4i __attribute__((vector_size(16)));
typedef __attribute__((address_space(1))) v4i gas_v4i;
typedef __attribute__((address_space(3))) v4i las_v4i;

static __device__ inline v16bf join16(v8bf lo, v8bf hi) {
  return __builtin_shufflevector(lo, hi, 0, 1, 2, 3, 4, 5, 6, 7, 8, 9, 10, 11, 12, 13, 14, 15);
}

// ---- async global -> LDS copy (16 bytes per lane), ASYNCcnt-tracked ----
static __device__ inline void async_cp16(bf16* ldst, const bf16* gsrc) {
#if __has_builtin(__builtin_amdgcn_global_load_async_to_lds_b128)
  __builtin_amdgcn_global_load_async_to_lds_b128((gas_v4i*)gsrc, (las_v4i*)ldst, 0, 0);
#else
  unsigned loff = (unsigned)(unsigned long long)(__attribute__((address_space(3))) bf16*)ldst;
  unsigned long long ga = (unsigned long long)gsrc;
  asm volatile("global_load_async_to_lds_b128 %0, %1, off" ::"v"(loff), "v"(ga) : "memory");
#endif
}

static __device__ inline void async_wait0() {
#if __has_builtin(__builtin_amdgcn_s_wait_asynccnt)
  __builtin_amdgcn_s_wait_asynccnt(0);
#else
  asm volatile("s_wait_asynccnt 0x0" ::: "memory");
#endif
}

// A fragment: 16(M) x 32(K) bf16 from row-major [., lda]; ISA 7.12.2 layout.
static __device__ inline v16bf load_frag_a(const bf16* __restrict__ base, int row0, int k0,
                                           int lda, int rmax, int lane) {
  int r = row0 + (lane & 15);
  r = (r > rmax) ? rmax : r;
  int kb = k0 + ((lane & 16) ? 8 : 0);
  const bf16* p = base + (long)r * lda + kb;
  v8bf lo = *(const v8bf*)(p);
  v8bf hi = *(const v8bf*)(p + 16);
  return join16(lo, hi);
}

// B fragment: 32(K) x 16(N) bf16 from N-major storage Bt[n][k], leading dim ldb.
static __device__ inline v16bf load_frag_b(const bf16* __restrict__ bt, int n0, int k0,
                                           int ldb, int nmax, int lane) {
  int n = n0 + (lane & 15);
  n = (n > nmax) ? nmax : n;
  int kb = k0 + ((lane & 16) ? 8 : 0);
  const bf16* p = bt + (long)n * ldb + kb;
  v8bf lo = *(const v8bf*)(p);
  v8bf hi = *(const v8bf*)(p + 16);
  return join16(lo, hi);
}

// B fragment from an LDS tile stored N-major with ldb = 32.
static __device__ inline v16bf load_frag_b_lds(const bf16* bs, int nloc, int lane) {
  int n = nloc + (lane & 15);
  int kb = (lane & 16) ? 8 : 0;
  const bf16* p = bs + n * 32 + kb;
  return join16(*(const v8bf*)p, *(const v8bf*)(p + 16));
}

static __device__ inline v8f wmma_bf16(v16bf a, v16bf b, v8f c) {
  return __builtin_amdgcn_wmma_f32_16x16x32_bf16(false, a, false, b, (short)0, c, false, false);
}

// ---------------- weight convert + transpose: Wt[n][k] = bf16(W[k][n]) ----------------
__global__ __launch_bounds__(256) void transpose_cvt_kernel(const float* __restrict__ W,
                                                            bf16* __restrict__ Wt,
                                                            int K, int N) {
  int idx = blockIdx.x * 256 + threadIdx.x;
  if (idx >= K * N) return;
  int n = idx / K;
  int k = idx - n * K;
  Wt[idx] = (bf16)W[(long)k * N + n];
}

// ---------------- layernorm (f32 in) -> bf16 out ----------------
__global__ __launch_bounds__(256) void ln_bf16_kernel(const float* __restrict__ x,
                                                      const float* __restrict__ g,
                                                      const float* __restrict__ b,
                                                      bf16* __restrict__ out) {
  int row = blockIdx.x;
  const float* xr = x + (long)row * EMBED;
  float v0 = xr[threadIdx.x];
  float v1 = xr[threadIdx.x + 256];
  float v2 = xr[threadIdx.x + 512];
  float s1 = v0 + v1 + v2;
  float s2 = v0 * v0 + v1 * v1 + v2 * v2;
  __shared__ float red1[256];
  __shared__ float red2[256];
  red1[threadIdx.x] = s1;
  red2[threadIdx.x] = s2;
  __syncthreads();
  for (int off = 128; off > 0; off >>= 1) {
    if (threadIdx.x < (unsigned)off) {
      red1[threadIdx.x] += red1[threadIdx.x + off];
      red2[threadIdx.x] += red2[threadIdx.x + off];
    }
    __syncthreads();
  }
  float mean = red1[0] * (1.0f / EMBED);
  float var = red2[0] * (1.0f / EMBED) - mean * mean;
  float rstd = rsqrtf(var + 1e-5f);
  bf16* orow = out + (long)row * EMBED;
  for (int i = threadIdx.x; i < EMBED; i += 256)
    orow[i] = (bf16)((xr[i] - mean) * rstd * g[i] + b[i]);
}

// ---------------- row sum of squares (bf16 in, f32 out) ----------------
__global__ __launch_bounds__(256) void rowsumsq_kernel(const bf16* __restrict__ q,
                                                       float* __restrict__ outv) {
  int row = blockIdx.x;
  const bf16* p = q + (long)row * EMBED;
  float s = 0.f;
  for (int i = threadIdx.x; i < EMBED; i += 256) {
    float f = (float)p[i];
    s += f * f;
  }
  __shared__ float red[256];
  red[threadIdx.x] = s;
  __syncthreads();
  for (int off = 128; off > 0; off >>= 1) {
    if (threadIdx.x < (unsigned)off) red[threadIdx.x] += red[threadIdx.x + off];
    __syncthreads();
  }
  if (threadIdx.x == 0) outv[row] = red[0];
}

// ---------------- WMMA GEMM: C[M,N] = A[M,K](bf16,row-major) x Bt[N,K](bf16,N-major) ----------------
// Block tile 128M x 128N, 8 waves (4 M-groups x 2 N-groups), wave tile 32x64.
// B tiles double-buffered in LDS via async global->LDS copies; A fragments double-buffered
// in registers so WMMAs overlap the next chunk's loads.
// EPI 0: C -> bf16 ; EPI 1: gelu(C+bias) -> bf16 ; EPI 2: C+bias+resid -> f32
template <int EPI>
__global__ __launch_bounds__(256) void gemm_bf16_kernel(const bf16* __restrict__ A,
                                                        const bf16* __restrict__ Bt,
                                                        const float* __restrict__ bias,
                                                        const float* __restrict__ resid,
                                                        bf16* __restrict__ Obf,
                                                        float* __restrict__ Of,
                                                        int M, int N, int K) {
  __shared__ bf16 Bs[2][128 * 32];  // 2 x 8KB
  int lane = threadIdx.x & 31;
  int wave = threadIdx.x >> 5;
  int wm = (wave & 3) * 32;
  int wn = (wave >> 2) * 64;
  int row0 = blockIdx.y * 128 + wm;
  int nblk = blockIdx.x * 128;

  auto stage = [&](int k0, int buf) {
    int r = threadIdx.x >> 1;         // 0..127 : tile row (n)
    int c = (threadIdx.x & 1) * 16;   // 0 or 16 : k offset
    int nrow = nblk + r;
    nrow = (nrow < N) ? nrow : N - 1;
    const bf16* src = Bt + (long)nrow * K + k0 + c;
    bf16* dst = &Bs[buf][r * 32 + c];
    async_cp16(dst, src);
    async_cp16(dst + 8, src + 8);
  };

  int nk = K >> 5;
  stage(0, 0);
  v16bf a0 = load_frag_a(A, row0, 0, K, M - 1, lane);
  v16bf a1 = load_frag_a(A, row0 + 16, 0, K, M - 1, lane);
  v8f acc[8] = {};
  async_wait0();
  __syncthreads();

  for (int kc = 0; kc < nk; ++kc) {
    bool more = (kc + 1) < nk;
    if (more) stage((kc + 1) * 32, (kc + 1) & 1);  // async: overlaps the WMMAs below
    v16bf na0, na1;
    if (more) {  // register-prefetch next A fragments
      na0 = load_frag_a(A, row0, (kc + 1) * 32, K, M - 1, lane);
      na1 = load_frag_a(A, row0 + 16, (kc + 1) * 32, K, M - 1, lane);
    }
    const bf16* bs = &Bs[kc & 1][0];
#pragma unroll
    for (int t = 0; t < 4; ++t) {
      v16bf b = load_frag_b_lds(bs, wn + t * 16, lane);
      acc[t] = wmma_bf16(a0, b, acc[t]);
      acc[4 + t] = wmma_bf16(a1, b, acc[4 + t]);
    }
    if (more) {
      a0 = na0;
      a1 = na1;
      async_wait0();
      __syncthreads();
    }
  }

  int cb = nblk + wn + (lane & 15);
  int rb = row0 + ((lane & 16) ? 8 : 0);
#pragma unroll
  for (int half = 0; half < 2; ++half) {
#pragma unroll
    for (int t = 0; t < 4; ++t) {
      int col = cb + t * 16;
#pragma unroll
      for (int r = 0; r < 8; ++r) {
        int row = rb + half * 16 + r;
        if (row >= M) continue;
        float v = acc[half * 4 + t][r];
        if (EPI >= 1) v += bias[col];
        if (EPI == 1) v = 0.5f * v * (1.0f + erff(v * 0.70710678118654752f));
        if (EPI == 2) {
          Of[(long)row * N + col] = resid[(long)row * N + col] + v;
        } else {
          Obf[(long)row * N + col] = (bf16)v;
        }
      }
    }
  }
}

// ---------------- L2-distance attention, one (batch, 16-query tile) per block ----------------
// scores = 2*Q.K^T - |q|^2 - |k|^2 ; softmax over keys ; out = P @ V ; xa = x + out
__global__ __launch_bounds__(256) void attn_kernel(const float* __restrict__ x,
                                                   const bf16* __restrict__ Q,
                                                   const bf16* __restrict__ Km,
                                                   const bf16* __restrict__ V,
                                                   const float* __restrict__ qsq,
                                                   const float* __restrict__ ksq,
                                                   float* __restrict__ xa) {
  extern __shared__ char smem[];
  float* sc = (float*)smem;                              // [16][SP] f32 scores
  bf16* pr = (bf16*)(smem + 16 * SP * 4);                // [16][SP] bf16 probs
  bf16* vt = (bf16*)(smem + 16 * SP * 4 + 16 * SP * 2);  // [EMBED][32] transposed V chunk

  int lane = threadIdx.x & 31;
  int wave = threadIdx.x >> 5;
  int b = blockIdx.y;
  int q0 = blockIdx.x * 16;
  const bf16* Qb = Q + (long)b * SEQ * EMBED;
  const bf16* Kb = Km + (long)b * SEQ * EMBED;
  const bf16* Vb = V + (long)b * SEQ * EMBED;
  const float* qsqb = qsq + b * SEQ;
  const float* ksqb = ksq + b * SEQ;

  // ---- phase 1: score tiles via WMMA, B fragments double-buffered ----
  for (int kt = wave; kt < SP / 16; kt += 8) {
    v8f acc = {};
    v16bf bb = load_frag_b(Kb, kt * 16, 0, EMBED, SEQ - 1, lane);
    for (int d = 0; d < EMBED; d += 32) {
      v16bf a = load_frag_a(Qb, q0, d, EMBED, SEQ - 1, lane);  // loop-invariant in kt, yet reloaded for every kt
      v16bf cur = bb;
      if (d + 32 < EMBED) bb = load_frag_b(Kb, kt * 16, d + 32, EMBED, SEQ - 1, lane);
      acc = wmma_bf16(a, cur, acc);
    }
    int kl = lane & 15;
    int kg = kt * 16 + kl;
    float kss = ksqb[kg < SEQ ? kg : SEQ - 1];
#pragma unroll
    for (int r = 0; r < 8; ++r) {
      int ql = r + ((lane & 16) ? 8 : 0);
      int qg = q0 + ql;
      float qss = qsqb[qg < SEQ ? qg : SEQ - 1];
      float s = 2.0f * acc[r] - qss - kss;
      if (kg >= SEQ) s = -1e30f;
      sc[ql * SP + kt * 16 + kl] = s;
    }
  }
  __syncthreads();

  // ---- phase 2: softmax (2 rows per wave) -> bf16 probs ----
  for (int qr = wave * 2; qr < wave * 2 + 2; ++qr) {
    float m = -1e30f;
    for (int i = lane; i < SP; i += 32) m = fmaxf(m, sc[qr * SP + i]);
    for (int off = 16; off > 0; off >>= 1) m = fmaxf(m, __shfl_xor(m, off, 32));
    float s = 0.f;
    for (int i = lane; i < SP; i += 32) {
      float e = __expf(sc[qr * SP + i] - m);
      sc[qr * SP + i] = e;
      s += e;
    }
    for (int off = 16; off > 0; off >>= 1) s += __shfl_xor(s, off, 32);
    float inv = 1.0f / s;
    for (int i = lane; i < SP; i += 32) pr[qr * SP + i] = (bf16)(sc[qr * SP + i] * inv);
  }
  __syncthreads();

  // ---- phase 3: out = P @ V, staging each 32-key V chunk transposed into LDS ----
  v8f oacc[6] = {};
  for (int kc = 0; kc < SP; kc += 32) {
    for (int i = threadIdx.x; i < 32 * EMBED; i += 256) {
      int key = i / EMBED;
      int d = i - key * EMBED;
      int kg = kc + key;
      kg = (kg < SEQ) ? kg : SEQ - 1;
      vt[d * 32 + key] = Vb[(long)kg * EMBED + d];
    }
    __syncthreads();
    v16bf a;
    {
      int ql = lane & 15;
      int kb = kc + ((lane & 16) ? 8 : 0);
      const bf16* p = pr + ql * SP + kb;
      a = join16(*(const v8bf*)p, *(const v8bf*)(p + 16));
    }
#pragma unroll
    for (int t = 0; t < 6; ++t) {
      v16bf bb = load_frag_b_lds(vt, wave * 96 + t * 16, lane);
      oacc[t] = wmma_bf16(a, bb, oacc[t]);
    }
    __syncthreads();
  }

  // ---- epilogue: xa = x + attn_out ----
#pragma unroll
  for (int t = 0; t < 6; ++t) {
    int col = wave * 96 + t * 16 + (lane & 15);
#pragma unroll
    for (int r = 0; r < 8; ++r) {
      int ql = r + ((lane & 16) ? 8 : 0);
      int qg = q0 + ql;
      if (qg < SEQ) {
        long idx = ((long)b * SEQ + qg) * EMBED + col;
        xa[idx] = x[idx] + oacc[t][r];
      }
    }
  }
}

extern "C" void kernel_launch(void* const* d_in, const int* in_sizes, int n_in,
                              void* d_out, int out_size, void* d_ws, size_t ws_size,
                              hipStream_t stream) {
  (void)in_sizes; (void)n_in; (void)out_size; (void)ws_size;
  const float* x    = (const float*)d_in[0];
  const float* Wq   = (const float*)d_in[1];
  const float* Wk   = (const float*)d_in[2];
  const float* Wv   = (const float*)d_in[3];
  const float* ln1g = (const float*)d_in[4];
  const float* ln1b = (const float*)d_in[5];
  const float* ln2g = (const float*)d_in[6];
  const float* ln2b = (const float*)d_in[7];
  const float* W1   = (const float*)d_in[8];
  const float* b1   = (const float*)d_in[9];
  const float* W2   = (const float*)d_in[10];
  const float* b2   = (const float*)d_in[11];
  float* out = (float*)d_out;

  char* w = (char*)d_ws;
  auto carve = [&](size_t bytes) {
    char* p = w;
    w += (bytes + 255) & ~(size_t)255;
    return p;
  };
  bf16* WqT = (bf16*)carve((size_t)EMBED * EMBED * 2);
  bf16* WkT = (bf16*)carve((size_t)EMBED * EMBED * 2);
  bf16* WvT = (bf16*)carve((size_t)EMBED * EMBED * 2);
  bf16* W1T = (bf16*)carve((size_t)EMBED * HIDDEN * 2);
  bf16* W2T = (bf16*)carve((size_t)HIDDEN * EMBED * 2);
  bf16* h1  = (bf16*)carve((size_t)NTOK * EMBED * 2);  // h1,Q,K,V contiguous;
  bf16* Qb  = (bf16*)carve((size_t)NTOK * EMBED * 2);  // reused as the GELU
  bf16* Kb  = (bf16*)carve((size_t)NTOK * EMBED * 2);  // activation buffer
  bf16* Vb  = (bf16*)carve((size_t)NTOK * EMBED * 2);
  bf16* act = h1;  // [NTOK, HIDDEN] bf16 == 4 * [NTOK, EMBED] bf16
  float* qsq = (float*)carve((size_t)NTOK * 4);
  float* ksq = (float*)carve((size_t)NTOK * 4);
  float* xa  = (float*)carve((size_t)NTOK * EMBED * 4);
  bf16* h2   = (bf16*)carve((size_t)NTOK * EMBED * 2);

  {
    int ne = EMBED * EMBED;
    transpose_cvt_kernel<<<(ne + 255) / 256, 256, 0, stream>>>(Wq, WqT, EMBED, EMBED);
    transpose_cvt_kernel<<<(ne + 255) / 256, 256, 0, stream>>>(Wk, WkT, EMBED, EMBED);
    transpose_cvt_kernel<<<(ne + 255) / 256, 256, 0, stream>>>(Wv, WvT, EMBED, EMBED);
    int nh = EMBED * HIDDEN;
    transpose_cvt_kernel<<<(nh + 255) / 256, 256, 0, stream>>>(W1, W1T, EMBED, HIDDEN);
    transpose_cvt_kernel<<<(nh + 255) / 256, 256, 0, stream>>>(W2, W2T, HIDDEN, EMBED);
  }

  ln_bf16_kernel<<<NTOK, 256, 0, stream>>>(x, ln1g, ln1b, h1);

  {
    dim3 grid(EMBED / 128, (NTOK + 127) / 128);
    gemm_bf16_kernel<0><<<grid, 256, 0, stream>>>(h1, WqT, nullptr, nullptr, Qb, nullptr, NTOK, EMBED, EMBED);
    gemm_bf16_kernel<0><<<grid, 256, 0, stream>>>(h1, WkT, nullptr, nullptr, Kb, nullptr, NTOK, EMBED, EMBED);
    gemm_bf16_kernel<0><<<grid, 256, 0, stream>>>(h1, WvT, nullptr, nullptr, Vb, nullptr, NTOK, EMBED, EMBED);
  }

  rowsumsq_kernel<<<NTOK, 256, 0, stream>>>(Qb, qsq);
  rowsumsq_kernel<<<NTOK, 256, 0, stream>>>(Kb, ksq);

  {
    dim3 grid((SEQ + 15) / 16, BATCH);
    size_t shmem = (size_t)16 * SP * 4 + (size_t)16 * SP * 2 + (size_t)EMBED * 32 * 2;
    attn_kernel<<<grid, 256, shmem, stream>>>(x, Qb, Kb, Vb, qsq, ksq, xa);
  }

  ln_bf16_kernel<<<NTOK, 256, 0, stream>>>(xa, ln2g, ln2b, h2);

  {
    dim3 grid(HIDDEN / 128, (NTOK + 127) / 128);
    gemm_bf16_kernel<1><<<grid, 256, 0, stream>>>(h2, W1T, b1, nullptr, act, nullptr, NTOK, HIDDEN, EMBED);
  }

  {
    dim3 grid(EMBED / 128, (NTOK + 127) / 128);
    gemm_bf16_kernel<2><<<grid, 256, 0, stream>>>(act, W2T, b2, xa, nullptr, out, NTOK, EMBED, HIDDEN);
  }
}
